// AttentionBlock_31782757990986
// MI455X (gfx1250) — matrix operand order repaired and checked by reading, not compiled, not run on hardware
//
#include <hip/hip_runtime.h>
#include <hip/hip_bf16.h>

// ---------------------------------------------------------------------------
// AttentionBlock for MI455X (gfx1250, wave32, WMMA f32_16x16x32_f16)
// Pipeline: instnorm1(+transpose) -> qkv GEMM -> per-(b,pix,head) attention
//           -> instnorm2 (in place) -> out-proj GEMM (+gamma residual)
// All matmuls use v_wmma_f32_16x16x32_f16 with f32 accumulation.
// ---------------------------------------------------------------------------

#define HE   12
#define HD   64
#define NTOK 64      // N tokens
#define EMB  768
#define PIX  256     // H*W
#define BN   128     // B*N
#define O3   2304    // 3*EMB
#define EPSN 1e-5f

typedef __attribute__((ext_vector_type(16))) _Float16 v16h;
typedef __attribute__((ext_vector_type(8)))  float    v8f;

__device__ __forceinline__ v8f vzero() {
  v8f z = {0.f,0.f,0.f,0.f,0.f,0.f,0.f,0.f};
  return z;
}

// A fragment: 16x32 (MxK). Lane l holds row M=(l&15); half slots
// [0..7] = K kb..kb+7, [8..15] = K kb+16..kb+23 with kb = (l&16)?8:0.
// Source row-major with row stride `ldh` halves (must be mult of 8).
__device__ __forceinline__ v16h frag_a(const _Float16* p0, int ldh, int lane) {
  const _Float16* p = p0 + (size_t)(lane & 15) * ldh + ((lane & 16) ? 8 : 0);
  union { v16h v; float4 q[2]; } u;
  u.q[0] = *(const float4*)(p);
  u.q[1] = *(const float4*)(p + 16);
  return u.v;
}

// B fragment: 32x16 (KxN). Lane l holds column N=(l&15); half slots
// [0..7] = K kb..kb+7, [8..15] = K kb+16..kb+23 with kb = (l&16)?8:0.
// Source is B^T row-major (row n = column n of B), row stride `ldh`.
__device__ __forceinline__ v16h frag_b(const _Float16* p0, int ldh, int lane) {
  const _Float16* p = p0 + (size_t)(lane & 15) * ldh + ((lane & 16) ? 8 : 0);
  union { v16h v; float4 q[2]; } u;
  u.q[0] = *(const float4*)(p);
  u.q[1] = *(const float4*)(p + 16);
  return u.v;
}

#define WMMA_F16(a, b, c) \
  __builtin_amdgcn_wmma_f32_16x16x32_f16(false, (a), false, (b), (short)0, (c), false, false)

// ---------------------------------------------------------------------------
// fp32 -> fp16 weight conversion
// ---------------------------------------------------------------------------
__global__ void __launch_bounds__(256) cvt_f32_f16(const float* __restrict__ s,
                                                   _Float16* __restrict__ d, int n) {
  int i = blockIdx.x * 256 + threadIdx.x;
  if (i < n) d[i] = (_Float16)s[i];
}

// ---------------------------------------------------------------------------
// InstanceNorm over pix for x[bn][c][pix] (f32), write transposed f16
// yT[bn][pix][c]. One WG = (bn, 64 channels); 256 thr = 4 thr/channel.
// ---------------------------------------------------------------------------
__global__ void __launch_bounds__(256) inorm1_to_yT(const float* __restrict__ x,
                                                    const float* __restrict__ w,
                                                    const float* __restrict__ b,
                                                    _Float16* __restrict__ yT) {
  const int bn = blockIdx.y;
  const int c0 = blockIdx.x * 64;
  const int t  = threadIdx.x;
  __shared__ _Float16 tile[PIX * 72];   // [pix][64ch] padded stride 72
  __shared__ float rs[256], rq[256], mus[64], rvs[64];

  const int cl = t >> 2, qd = t & 3;
  const int c  = c0 + cl;
  const float* src = x + ((size_t)bn * EMB + c) * PIX + qd * 64;

  float s = 0.f, s2 = 0.f;
  for (int i = 0; i < 64; i += 4) {
    float4 v = *(const float4*)(src + i);
    s  += v.x + v.y + v.z + v.w;
    s2 += v.x*v.x + v.y*v.y + v.z*v.z + v.w*v.w;
  }
  rs[t] = s; rq[t] = s2;
  __syncthreads();
  if (qd == 0) {
    float ts = rs[t] + rs[t+1] + rs[t+2] + rs[t+3];
    float tq = rq[t] + rq[t+1] + rq[t+2] + rq[t+3];
    float mu = ts * (1.f / PIX);
    float var = tq * (1.f / PIX) - mu * mu;
    mus[cl] = mu;
    rvs[cl] = rsqrtf(var + EPSN);
  }
  __syncthreads();
  const float mu = mus[cl], rv = rvs[cl];
  const float wc = w[c], bc = b[c];
  for (int i = 0; i < 64; i += 4) {
    float4 v = *(const float4*)(src + i);
    int p = qd * 64 + i;
    tile[(p+0)*72 + cl] = (_Float16)((v.x - mu) * rv * wc + bc);
    tile[(p+1)*72 + cl] = (_Float16)((v.y - mu) * rv * wc + bc);
    tile[(p+2)*72 + cl] = (_Float16)((v.z - mu) * rv * wc + bc);
    tile[(p+3)*72 + cl] = (_Float16)((v.w - mu) * rv * wc + bc);
  }
  __syncthreads();
  // each thread writes one pix row: 64 contiguous halves
  _Float16* dst = yT + ((size_t)bn * PIX + t) * EMB + c0;
  const _Float16* srow = tile + t * 72;
  for (int i = 0; i < 64; i += 8)
    *(float4*)(dst + i) = *(const float4*)(srow + i);
}

// ---------------------------------------------------------------------------
// qkv GEMM: per bn, C[pix][o] = yT[pix][:] . w_qkv[o][:] + b_qkv[o]
// M=pix (tile 64), N=o (tile 128, 32/wave), K=768. Output qkvt[bn][pix][o] f16.
// ---------------------------------------------------------------------------
__global__ void __launch_bounds__(128) gemm_qkv(const _Float16* __restrict__ yT,
                                                const _Float16* __restrict__ wq,
                                                const float* __restrict__ bq,
                                                _Float16* __restrict__ qkvt) {
  const int bn = blockIdx.z;
  const int m0 = blockIdx.y * 64;    // pix tile
  const int n0 = blockIdx.x * 128;   // o tile
  const int wv = threadIdx.x >> 5, lane = threadIdx.x & 31;

  const _Float16* Abase = yT + ((size_t)bn * PIX + m0) * EMB;
  const _Float16* Bbase = wq + (size_t)(n0 + wv * 32) * EMB;

  v8f acc[4][2];
  for (int i = 0; i < 4; ++i) { acc[i][0] = vzero(); acc[i][1] = vzero(); }

  for (int k = 0; k < EMB; k += 32) {
    v16h bf0 = frag_b(Bbase + k, EMB, lane);
    v16h bf1 = frag_b(Bbase + (size_t)16 * EMB + k, EMB, lane);
    for (int mf = 0; mf < 4; ++mf) {
      v16h af = frag_a(Abase + (size_t)mf * 16 * EMB + k, EMB, lane);
      acc[mf][0] = WMMA_F16(af, bf0, acc[mf][0]);
      acc[mf][1] = WMMA_F16(af, bf1, acc[mf][1]);
    }
  }
  const int mh = (lane & 16) ? 8 : 0;
  for (int nf = 0; nf < 2; ++nf) {
    const int o = n0 + wv * 32 + nf * 16 + (lane & 15);
    const float bias = bq[o];
    for (int mf = 0; mf < 4; ++mf)
      for (int r = 0; r < 8; ++r) {
        const int pix = m0 + mf * 16 + r + mh;
        qkvt[((size_t)bn * PIX + pix) * O3 + o] = (_Float16)(acc[mf][nf][r] + bias);
      }
  }
}

// ---------------------------------------------------------------------------
// Attention: one WG (128 thr = 4 waves) per (b, pix, head).
// LDS: q,k (LayerNormed), v^T, scores (f32 softmax), probs (f16).
// S = qk^T/8 + bias, O = softmax(S) @ V ; write ot[bn][pix][head*64+d] f16.
// ---------------------------------------------------------------------------
__global__ void __launch_bounds__(128) attn_kernel(const _Float16* __restrict__ qkvt,
                                                   const float* __restrict__ qnw,
                                                   const float* __restrict__ qnb,
                                                   const float* __restrict__ knw,
                                                   const float* __restrict__ knb,
                                                   const float* __restrict__ rel,
                                                   _Float16* __restrict__ ot) {
  const int h = blockIdx.x, pix = blockIdx.y, b = blockIdx.z;
  const int t = threadIdx.x, wv = t >> 5, lane = t & 31;

  __shared__ _Float16 qs[NTOK * 72];
  __shared__ _Float16 ks[NTOK * 72];
  __shared__ _Float16 vT[NTOK * 72];
  __shared__ _Float16 Ps[NTOK * 72];
  __shared__ float    Ss[NTOK * 65];
  __shared__ float    bias_s[127 * HE];

  for (int i = t; i < 127 * HE; i += 128) bias_s[i] = rel[i];

  const int n = t & 63;
  const _Float16* src = qkvt + ((size_t)(b * NTOK + n) * PIX + pix) * O3 + h * 192;
  if (t < 64) {
    for (int j = 0; j < 64; j += 8) {
      *(float4*)(qs + n * 72 + j) = *(const float4*)(src + j);
      *(float4*)(ks + n * 72 + j) = *(const float4*)(src + 64 + j);
    }
  } else {
    for (int d = 0; d < 64; ++d) vT[d * 72 + n] = src[128 + d];  // transpose V
  }
  __syncthreads();

  // LayerNorm over head_dim: threads 0-63 -> q rows, 64-127 -> k rows
  {
    _Float16* row = (t < 64) ? (qs + n * 72) : (ks + n * 72);
    const float* lw = (t < 64) ? qnw : knw;
    const float* lb = (t < 64) ? qnb : knb;
    float s = 0.f, s2 = 0.f;
    for (int d = 0; d < 64; ++d) { float v = (float)row[d]; s += v; s2 += v * v; }
    const float mu = s * (1.f / 64.f);
    const float rv = rsqrtf(s2 * (1.f / 64.f) - mu * mu + EPSN);
    for (int d = 0; d < 64; ++d) {
      float v = (float)row[d];
      row[d] = (_Float16)((v - mu) * rv * lw[d] + lb[d]);
    }
  }
  __syncthreads();

  // S = q @ k^T * scale + rel bias ; wave wv owns 16 query rows
  const float scale = 0.125f;  // 64^-0.5
  {
    const _Float16* arow = qs + wv * 16 * 72;
    for (int nf = 0; nf < 4; ++nf) {
      v8f acc = vzero();
      for (int kk = 0; kk < 2; ++kk) {
        v16h a  = frag_a(arow + kk * 32, 72, lane);
        v16h bb = frag_b(ks + nf * 16 * 72 + kk * 32, 72, lane);
        acc = WMMA_F16(a, bb, acc);
      }
      const int mh = (lane & 16) ? 8 : 0;
      const int nn = nf * 16 + (lane & 15);
      for (int r = 0; r < 8; ++r) {
        const int m = wv * 16 + r + mh;
        Ss[m * 65 + nn] = acc[r] * scale + bias_s[(m - nn + 63) * HE + h];
      }
    }
  }
  __syncthreads();

  // row softmax (f32), emit f16 probabilities
  if (t < 64) {
    float* row = Ss + t * 65;
    float mx = row[0];
    for (int j = 1; j < 64; ++j) mx = fmaxf(mx, row[j]);
    float sum = 0.f;
    for (int j = 0; j < 64; ++j) { float e = __expf(row[j] - mx); row[j] = e; sum += e; }
    const float inv = 1.f / sum;
    _Float16* pr = Ps + t * 72;
    for (int j = 0; j < 64; ++j) pr[j] = (_Float16)(row[j] * inv);
  }
  __syncthreads();

  // O = P @ V  (B cols = columns of V = rows of vT)
  {
    const _Float16* arow = Ps + wv * 16 * 72;
    for (int nf = 0; nf < 4; ++nf) {
      v8f acc = vzero();
      for (int kk = 0; kk < 2; ++kk) {
        v16h a  = frag_a(arow + kk * 32, 72, lane);
        v16h bb = frag_b(vT + nf * 16 * 72 + kk * 32, 72, lane);
        acc = WMMA_F16(a, bb, acc);
      }
      const int mh = (lane & 16) ? 8 : 0;
      const int d  = nf * 16 + (lane & 15);
      for (int r = 0; r < 8; ++r) {
        const int m = wv * 16 + r + mh;  // query token
        ot[((size_t)(b * NTOK + m) * PIX + pix) * EMB + h * HD + d] = (_Float16)acc[r];
      }
    }
  }
}

// ---------------------------------------------------------------------------
// InstanceNorm2 in place on ot[bn][pix][c] f16 (stats over pix per channel).
// ---------------------------------------------------------------------------
__global__ void __launch_bounds__(256) inorm2_inplace(_Float16* __restrict__ ot,
                                                      const float* __restrict__ w,
                                                      const float* __restrict__ b) {
  const int bn = blockIdx.y;
  const int c0 = blockIdx.x * 64;
  const int t  = threadIdx.x;
  __shared__ _Float16 tile[PIX * 72];
  __shared__ float rs[256], rq[256], mus[64], rvs[64];

  _Float16* row = ot + ((size_t)bn * PIX + t) * EMB + c0;
  for (int i = 0; i < 64; i += 8)
    *(float4*)(tile + t * 72 + i) = *(const float4*)(row + i);
  __syncthreads();

  const int cl = t >> 2, qd = t & 3;
  float s = 0.f, s2 = 0.f;
  for (int p = qd * 64; p < qd * 64 + 64; ++p) {
    float v = (float)tile[p * 72 + cl];
    s += v; s2 += v * v;
  }
  rs[t] = s; rq[t] = s2;
  __syncthreads();
  if (qd == 0) {
    float ts = rs[t] + rs[t+1] + rs[t+2] + rs[t+3];
    float tq = rq[t] + rq[t+1] + rq[t+2] + rq[t+3];
    float mu = ts * (1.f / PIX);
    float var = tq * (1.f / PIX) - mu * mu;
    mus[cl] = mu;
    rvs[cl] = rsqrtf(var + EPSN);
  }
  __syncthreads();
  const float mu = mus[cl], rv = rvs[cl];
  const float wc = w[c0 + cl], bc = b[c0 + cl];
  for (int p = qd * 64; p < qd * 64 + 64; ++p) {
    float v = (float)tile[p * 72 + cl];
    tile[p * 72 + cl] = (_Float16)((v - mu) * rv * wc + bc);
  }
  __syncthreads();
  for (int i = 0; i < 64; i += 8)
    *(float4*)(row + i) = *(const float4*)(tile + t * 72 + i);
}

// ---------------------------------------------------------------------------
// Out projection: per bn, out[c][pix] = (w_out[c][:] . on[pix][:] + b_out[c])
//                                        * gamma[c] + x[bn][c][pix]
// M=c_out (tile 128, 32/wave), N=pix (tile 64), K=768. f32 output.
// ---------------------------------------------------------------------------
__global__ void __launch_bounds__(128) gemm_out(const _Float16* __restrict__ wo,
                                                const _Float16* __restrict__ on,
                                                const float* __restrict__ bo,
                                                const float* __restrict__ gm,
                                                const float* __restrict__ xin,
                                                float* __restrict__ out) {
  const int bn = blockIdx.z;
  const int c0 = blockIdx.y * 128;   // c_out tile
  const int p0 = blockIdx.x * 64;    // pix tile
  const int wv = threadIdx.x >> 5, lane = threadIdx.x & 31;

  const _Float16* Abase = wo + (size_t)(c0 + wv * 32) * EMB;
  const _Float16* Bbase = on + ((size_t)bn * PIX + p0) * EMB;

  v8f acc[2][4];
  for (int i = 0; i < 2; ++i)
    for (int j = 0; j < 4; ++j) acc[i][j] = vzero();

  for (int k = 0; k < EMB; k += 32) {
    v16h af0 = frag_a(Abase + k, EMB, lane);
    v16h af1 = frag_a(Abase + (size_t)16 * EMB + k, EMB, lane);
    for (int nf = 0; nf < 4; ++nf) {
      v16h bf = frag_b(Bbase + (size_t)nf * 16 * EMB + k, EMB, lane);
      acc[0][nf] = WMMA_F16(af0, bf, acc[0][nf]);
      acc[1][nf] = WMMA_F16(af1, bf, acc[1][nf]);
    }
  }
  const int mh = (lane & 16) ? 8 : 0;
  for (int mf = 0; mf < 2; ++mf)
    for (int r = 0; r < 8; ++r) {
      const int c = c0 + wv * 32 + mf * 16 + r + mh;
      const float g = gm[c], bb = bo[c];
      for (int nf = 0; nf < 4; ++nf) {
        const int pix = p0 + nf * 16 + (lane & 15);
        const size_t idx = ((size_t)bn * EMB + c) * PIX + pix;
        out[idx] = (acc[mf][nf][r] + bb) * g + xin[idx];
      }
    }
}

// ---------------------------------------------------------------------------
extern "C" void kernel_launch(void* const* d_in, const int* in_sizes, int n_in,
                              void* d_out, int out_size, void* d_ws, size_t ws_size,
                              hipStream_t stream) {
  const float* x     = (const float*)d_in[0];
  const float* n1w   = (const float*)d_in[1];
  const float* n1b   = (const float*)d_in[2];
  const float* wqkv  = (const float*)d_in[3];
  const float* bqkv  = (const float*)d_in[4];
  const float* qnw   = (const float*)d_in[5];
  const float* qnb   = (const float*)d_in[6];
  const float* knw   = (const float*)d_in[7];
  const float* knb   = (const float*)d_in[8];
  const float* rel   = (const float*)d_in[9];
  const float* n2w   = (const float*)d_in[10];
  const float* n2b   = (const float*)d_in[11];
  const float* wout  = (const float*)d_in[12];
  const float* bout  = (const float*)d_in[13];
  const float* gamma = (const float*)d_in[14];
  float* out = (float*)d_out;

  // workspace layout (halves): wq16 | wo16 | yT | qkvt ; ot aliases yT
  _Float16* ws   = (_Float16*)d_ws;
  _Float16* wq16 = ws;                                   // 2304*768
  _Float16* wo16 = wq16 + (size_t)O3 * EMB;              // 768*768
  _Float16* yT   = wo16 + (size_t)EMB * EMB;             // 128*256*768
  _Float16* qkvt = yT + (size_t)BN * PIX * EMB;          // 128*256*2304
  _Float16* ot   = yT;                                   // reuse: yT dead after qkv GEMM

  cvt_f32_f16<<<(O3 * EMB + 255) / 256, 256, 0, stream>>>(wqkv, wq16, O3 * EMB);
  cvt_f32_f16<<<(EMB * EMB + 255) / 256, 256, 0, stream>>>(wout, wo16, EMB * EMB);

  inorm1_to_yT<<<dim3(EMB / 64, BN), 256, 0, stream>>>(x, n1w, n1b, yT);

  gemm_qkv<<<dim3(O3 / 128, PIX / 64, BN), 128, 0, stream>>>(yT, wq16, bqkv, qkvt);

  attn_kernel<<<dim3(HE, PIX, 2), 128, 0, stream>>>(qkvt, qnw, qnb, knw, knb, rel, ot);

  inorm2_inplace<<<dim3(EMB / 64, BN), 256, 0, stream>>>(ot, n2w, n2b);

  gemm_out<<<dim3(PIX / 64, EMB / 128, BN), 128, 0, stream>>>(wo16, ot, bout, gamma, x, out);
}
